// MoE_LoRA_15968688406555
// MI455X (gfx1250) — matrix operand order repaired and checked by reading, not compiled, not run on hardware
//
#include <hip/hip_runtime.h>

#define B_   32
#define S_   4096
#define D_   1280
#define R_   64
#define E_   8
#define RC_  128           // 2R: concatenated rank
#define ALPHA_ 2.0f

typedef __attribute__((ext_vector_type(16))) __bf16 v16bf;
typedef __attribute__((ext_vector_type(2)))  __bf16 v2bf;
typedef __attribute__((ext_vector_type(8)))  float  v8f;
typedef __attribute__((ext_vector_type(4)))  unsigned int v4u;
typedef __attribute__((ext_vector_type(8)))  int v8i;
typedef __attribute__((ext_vector_type(4)))  int v4i;
typedef unsigned short u16;
typedef unsigned int   u32;
typedef unsigned long long u64;

#if __has_builtin(__builtin_amdgcn_tensor_load_to_lds) && \
    __has_builtin(__builtin_amdgcn_s_wait_tensorcnt)
#define HAVE_TDM 1
#else
#define HAVE_TDM 0
#endif

__device__ __forceinline__ u16 f2bf(float f) {
  u32 u = __float_as_uint(f);
  u32 r = (u + 0x7FFFu + ((u >> 16) & 1u)) >> 16;   // RNE
  return (u16)r;
}

// two f32 -> packed bf16 dword (hardware V_CVT_PK_BF16_F32 when available)
__device__ __forceinline__ u32 pk2bf(float a, float b) {
#if __has_builtin(__builtin_amdgcn_cvt_pk_bf16_f32)
  union { v2bf v; u32 u; } c;
  c.v = __builtin_amdgcn_cvt_pk_bf16_f32(a, b);
  return c.u;
#else
  return (u32)f2bf(a) | ((u32)f2bf(b) << 16);
#endif
}

// ---------------------------------------------------------------------------
// 1-D TDM copy: ndw dwords from contiguous global -> contiguous LDS.
// D# group0: count=1, lds_addr, global_addr, type=2. group1: data_size=4B,
// tensor_dim0 = tile_dim0 = ndw, 1-D (tile_dim1..=0).
// ---------------------------------------------------------------------------
__device__ __forceinline__ void tdm_issue_1d(u32 lds_off, const void* src, int ndw) {
#if HAVE_TDM
  u64 ga = (u64)(size_t)src;
  v4u g0;
  g0[0] = 1u;                                        // count=1
  g0[1] = lds_off;                                   // lds_addr (bytes)
  g0[2] = (u32)ga;                                   // global_addr[31:0]
  g0[3] = (u32)((ga >> 32) & 0x1FFFFFFu) | (2u << 30); // addr[56:32] | type=2
  v8i g1;
  g1[0] = (int)(2u << 16);                           // data_size = 4 bytes
  g1[1] = (int)(((u32)ndw & 0xFFFFu) << 16);         // tensor_dim0[15:0]
  g1[2] = (int)((u32)ndw >> 16);                     // tensor_dim0[31:16]
  g1[3] = (int)(((u32)ndw & 0xFFFFu) << 16);         // tile_dim0
  g1[4] = 0;                                         // tile_dim1/2 unused
  g1[5] = ndw;                                       // tensor_dim0_stride
  g1[6] = 0;
  g1[7] = 0;
  v4i z = {0, 0, 0, 0};
#if __clang_major__ >= 23
  v8i z8 = {0, 0, 0, 0, 0, 0, 0, 0};
  __builtin_amdgcn_tensor_load_to_lds(g0, g1, z, z, z8, 0);
#else
  __builtin_amdgcn_tensor_load_to_lds(g0, g1, z, z, 0);
#endif
#endif
}

__device__ __forceinline__ void stage_issue(u32 lds_off, u32* dst, const u32* src,
                                            int ndw, int tid, int wv) {
#if HAVE_TDM
  (void)dst; (void)tid;
  if (wv == 0) tdm_issue_1d(lds_off, src, ndw);
#else
  (void)lds_off; (void)wv;
  for (int i = tid; i < ndw / 4; i += 256)
    ((uint4*)dst)[i] = ((const uint4*)src)[i];
#endif
}

__device__ __forceinline__ void stage_wait(int wv) {
#if HAVE_TDM
  if (wv == 0) __builtin_amdgcn_s_wait_tensorcnt(0);
#else
  (void)wv;
#endif
}

union Frag { v16bf v; uint4 q[2]; };

// ---------------------------------------------------------------------------
// Prep: repack LoRA weights to bf16 so every LDS stage is a contiguous
// global region (TDM-friendly).
//   AcatBF tiled:  [e][dkb=40][r=128][dd=32]   (chunk = 8 KB = 2048 dw)
//   BpackBF tiled: [e][dcb=20][g=8][dd=64][j=8] dwords (chunk = 16 KB = 4096 dw)
//     dword (g,dd,j) = bf16 pair Bcat[d][2*rp], Bcat[d][2*rp+1], rp = g*8+j,
//     d = dcb*64+dd;  Bcat[d][r] = r<64 ? Bex[e][d][r] : Bgen[d][r-64]
// ---------------------------------------------------------------------------
__global__ __launch_bounds__(256) void lora_prep(
    const float* __restrict__ Aex, const float* __restrict__ Bex,
    const float* __restrict__ Agen, const float* __restrict__ Bgen,
    u16* __restrict__ AcatBF, u16* __restrict__ BpackBF)
{
  int idx = blockIdx.x * 256 + threadIdx.x;
  const int tot = E_ * RC_ * D_;
  if (idx >= tot) return;

  // A side
  {
    int d = idx % D_;
    int t = idx / D_;
    int r = t % RC_;
    int e = t / RC_;
    float v = (r < R_) ? Aex[((size_t)e * R_ + r) * D_ + d]
                       : Agen[(size_t)(r - R_) * D_ + d];
    AcatBF[(((size_t)e * 40 + (d >> 5)) * 128 + r) * 32 + (d & 31)] = f2bf(v);
  }
  // B side (same-sized index space)
  {
    int i  = idx & 1;
    int t2 = idx >> 1;
    int d  = t2 % D_;
    int t3 = t2 / D_;
    int rp = t3 % (RC_ / 2);
    int e  = t3 / (RC_ / 2);
    int rr = rp * 2 + i;
    float w = (rr < R_) ? Bex[((size_t)e * D_ + d) * R_ + rr]
                        : Bgen[(size_t)d * R_ + (rr - R_)];
    size_t o = ((((size_t)e * 20 + (d >> 6)) * 8 + (rp >> 3)) * 64 + (d & 63)) * 8
               + (rp & 7);
    BpackBF[o * 2 + i] = f2bf(w);
  }
}

// ---------------------------------------------------------------------------
// Main fused LoRA kernel: one block (8 waves) = 128 S-rows of one sample.
// Phase 1: h[128,128] = x_tile @ Acat^T   (bf16 WMMA, f32 acc)
// Phase 2: out_tile   = ALPHA * h @ Bcat^T
// Weight chunks double-buffered via TDM, overlapped with compute.
// ---------------------------------------------------------------------------
__global__ __launch_bounds__(256) void lora_main(
    const float* __restrict__ x, const int* __restrict__ label,
    const u16* __restrict__ AcatBF, const u32* __restrict__ BpackU,
    float* __restrict__ out)
{
  __shared__ __align__(16) unsigned char smem[64 * 1024];
  u16* hsm = (u16*)smem;                    // [128][128] bf16, 32 KB
  u16* xa  = (u16*)(smem + 32 * 1024);      // [128][32]  bf16, 8 KB   (phase 1)
  u16* ac0 = (u16*)(smem + 40 * 1024);      // Acat chunk buf 0, 8 KB  (phase 1)
  u16* ac1 = (u16*)(smem + 48 * 1024);      // Acat chunk buf 1, 8 KB  (phase 1)
  u32* bs0 = (u32*)(smem + 32 * 1024);      // Bpack chunk buf 0, 16 KB (phase 2)
  u32* bs1 = (u32*)(smem + 48 * 1024);      // Bpack chunk buf 1, 16 KB (phase 2)

  const int sb  = blockIdx.x;
  const int b   = blockIdx.y;
  const int tid = threadIdx.x;
  float* outb = out + ((size_t)b * S_ + (size_t)sb * 128) * D_;

  if (b == B_ - 1) {                        // reference zeroes last sample
    float4 z = make_float4(0.f, 0.f, 0.f, 0.f);
    float4* o4 = (float4*)outb;
    for (int i = tid; i < 128 * D_ / 4; i += 256) o4[i] = z;
    return;
  }

  const int e = label[b];
  const u16* Ae2 = AcatBF + (size_t)e * RC_ * D_;          // 40 chunks * 2048 dw
  const u32* Bp2 = BpackU + (size_t)e * (RC_ / 2) * D_;    // 20 chunks * 4096 dw
  const float* xb = x + ((size_t)b * S_ + (size_t)sb * 128) * D_;

  const int wv = tid >> 5;                  // wave -> M-tile
  const int ln = tid & 31;
  const int n  = ln & 15;
  const int kh = ln >> 4;                   // K-half select (16-bit layouts)

  const u32 lds_ac0 = (u32)(size_t)ac0;
  const u32 lds_ac1 = (u32)(size_t)ac1;
  const u32 lds_bs0 = (u32)(size_t)bs0;
  const u32 lds_bs1 = (u32)(size_t)bs1;

  const v8f vzero = {0.f, 0.f, 0.f, 0.f, 0.f, 0.f, 0.f, 0.f};
  v8f acc[8];
#pragma unroll
  for (int t = 0; t < 8; ++t) acc[t] = vzero;

  // ---------------- Phase 1: h = x @ Acat^T ----------------
  stage_issue(lds_ac0, (u32*)ac0, (const u32*)Ae2, 2048, tid, wv);   // preload

  for (int dkb = 0; dkb < 40; ++dkb) {
    const int dk = dkb * 32;
    // stage x chunk [128 x 32] f32 -> bf16 (overlaps in-flight TDM)
#pragma unroll
    for (int i = 0; i < 4; ++i) {
      int l = tid + i * 256;
      int row = l >> 3, c4 = l & 7;
      float4 v = *(const float4*)(xb + (size_t)row * D_ + dk + c4 * 4);
      uint2 pk;
      pk.x = pk2bf(v.x, v.y);
      pk.y = pk2bf(v.z, v.w);
      *(uint2*)(xa + row * 32 + c4 * 4) = pk;
    }
    stage_wait(wv);
    __syncthreads();
    if (dkb + 1 < 40) {                     // prefetch next chunk (other buffer)
      u32  nlds = (dkb & 1) ? lds_ac0 : lds_ac1;
      u32* ndst = (dkb & 1) ? (u32*)ac0 : (u32*)ac1;
      stage_issue(nlds, ndst, (const u32*)(Ae2 + (size_t)(dkb + 1) * 4096),
                  2048, tid, wv);
    }
    const u16* ac = (dkb & 1) ? ac1 : ac0;

    // A fragment: elems[i]=K(16*kh+i)
    Frag af;
    const uint4* ap = (const uint4*)(xa + (wv * 16 + n) * 32);
    af.q[0] = ap[2 * kh];
    af.q[1] = ap[2 * kh + 1];
#pragma unroll
    for (int nt = 0; nt < 8; ++nt) {        // 8 rank-tiles (N)
      Frag bfr;                             // B fragment: elems[i]=K(16*kh+i)
      const uint4* bp = (const uint4*)(ac + (nt * 16 + n) * 32 + kh * 16);
      bfr.q[0] = bp[0];
      bfr.q[1] = bp[1];
      acc[nt] = __builtin_amdgcn_wmma_f32_16x16x32_bf16(
          false, af.v, false, bfr.v, (short)0, acc[nt], false, false);
    }
    __syncthreads();
  }

  // h -> LDS bf16 (C layout: VGPR j -> M = j + 8*kh, N = nt*16 + n)
#pragma unroll
  for (int nt = 0; nt < 8; ++nt)
#pragma unroll
    for (int j = 0; j < 8; j += 2) {
      u32 w = pk2bf(acc[nt][j], acc[nt][j + 1]);
      int base = (wv * 16 + j + 8 * kh) * RC_ + nt * 16 + n;
      hsm[base]       = (u16)w;
      hsm[base + RC_] = (u16)(w >> 16);
    }
  __syncthreads();

  // ---------------- Phase 2: out = ALPHA * h @ Bcat^T ----------------
  stage_issue(lds_bs0, bs0, Bp2, 4096, tid, wv);                    // preload

  for (int dcb = 0; dcb < 20; ++dcb) {      // 64 output columns per chunk
    stage_wait(wv);
    __syncthreads();
    if (dcb + 1 < 20) {
      u32  nlds = (dcb & 1) ? lds_bs0 : lds_bs1;
      u32* ndst = (dcb & 1) ? bs0 : bs1;
      stage_issue(nlds, ndst, Bp2 + (size_t)(dcb + 1) * 4096, 4096, tid, wv);
    }
    const u32* bsm = (dcb & 1) ? bs1 : bs0;

    v8f acc2[4];
#pragma unroll
    for (int t = 0; t < 4; ++t) acc2[t] = vzero;

    const u16* hrow = hsm + (wv * 16 + n) * RC_;
#pragma unroll
    for (int kc = 0; kc < 4; ++kc) {        // K = 128 in chunks of 32
      Frag af2;
      af2.q[0] = *(const uint4*)(hrow + 32 * kc + 16 * kh);
      af2.q[1] = *(const uint4*)(hrow + 32 * kc + 16 * kh + 8);
      const int g = 2 * kc + kh;
#pragma unroll
      for (int nt = 0; nt < 4; ++nt) {      // 4 output-column tiles
        Frag bf2;
        const uint4* bq = (const uint4*)(bsm + ((size_t)(g * 64 + nt * 16 + n)) * 8);
        bf2.q[0] = bq[0];
        bf2.q[1] = bq[1];
        acc2[nt] = __builtin_amdgcn_wmma_f32_16x16x32_bf16(
            false, af2.v, false, bf2.v, (short)0, acc2[nt], false, false);
      }
    }
    // scaled f32 store (coalesced across lanes)
#pragma unroll
    for (int nt = 0; nt < 4; ++nt)
#pragma unroll
      for (int j = 0; j < 8; ++j)
        outb[(size_t)(wv * 16 + j + 8 * kh) * D_ + dcb * 64 + nt * 16 + n] =
            ALPHA_ * acc2[nt][j];
    __syncthreads();
  }
}

// ---------------------------------------------------------------------------
extern "C" void kernel_launch(void* const* d_in, const int* in_sizes, int n_in,
                              void* d_out, int out_size, void* d_ws, size_t ws_size,
                              hipStream_t stream) {
  const float* x    = (const float*)d_in[0];
  // d_in[1] = weight (unused by the reference computation)
  const float* Aex  = (const float*)d_in[2];
  const float* Bex  = (const float*)d_in[3];
  const float* Agen = (const float*)d_in[4];
  const float* Bgen = (const float*)d_in[5];
  const int*   lbl  = (const int*)d_in[6];
  float* out = (float*)d_out;

  u16* AcatBF  = (u16*)d_ws;                         // 2.62 MB
  u16* BpackBF = AcatBF + (size_t)E_ * RC_ * D_;     // 2.62 MB

  const int tot = E_ * RC_ * D_;
  lora_prep<<<(tot + 255) / 256, 256, 0, stream>>>(Aex, Bex, Agen, Bgen,
                                                   AcatBF, BpackBF);

  dim3 grid(S_ / 128, B_);
  lora_main<<<grid, 256, 0, stream>>>(x, lbl, AcatBF, (const u32*)BpackBF, out);
}
